// TFTMultiHeadAttention_5592047419485
// MI455X (gfx1250) — compile-verified, matrix operand order checked by reading, not run on hardware
//
#include <hip/hip_runtime.h>
#include <hip/hip_bf16.h>

// ---------------------------------------------------------------------------
// TFT MultiHeadAttention for gfx1250 (MI455X), wave32 + v_wmma_f32_16x16x32_f16
// H=8, B=16, S=1024, D_MODEL=512, D_K=64
// ---------------------------------------------------------------------------

typedef _Float16 v16h __attribute__((ext_vector_type(16)));
typedef _Float16 h8   __attribute__((ext_vector_type(8)));
typedef float    v8f  __attribute__((ext_vector_type(8)));

#define HN   8
#define BN   16
#define SN   1024
#define DM   512
#define DKN  64
#define MROWS (BN * SN)        // 16384
#define NEGV (-1.0e9f)

// ---- WMMA fragment loaders --------------------------------------------------
// A-matrix 16x32 f16 layout (ISA 7.12.2): lane<16 -> row = lane, halves[0..7] =
// K {0..7}, halves[8..15] = K {16..23}; lane>=16 -> row = lane-16, K {8..15},
// K {24..31}.  B-matrix (KxN) is symmetric with lane = column, so loading from
// a [N][K] buffer uses identical addressing.  Chunks are 16B-aligned -> b128.
__device__ __forceinline__ v16h load_frag_h(const _Float16* base, int row0,
                                            int rs, int kc) {
  int lane = threadIdx.x & 31;
  const _Float16* p =
      base + (size_t)(row0 + (lane & 15)) * rs + kc + ((lane >> 4) << 3);
  v16h r;
  ((h8*)&r)[0] = *(const h8*)(p);
  ((h8*)&r)[1] = *(const h8*)(p + 16);
  return r;
}

// Same fragment sourced from an f32 row-major buffer (converted on load).
__device__ __forceinline__ v16h load_frag_f32(const float* base, int row0,
                                              int rs, int kc) {
  int lane = threadIdx.x & 31;
  const float* p =
      base + (size_t)(row0 + (lane & 15)) * rs + kc + ((lane >> 4) << 3);
  v16h r;
#pragma unroll
  for (int i = 0; i < 8; ++i) r[i] = (_Float16)p[i];
#pragma unroll
  for (int i = 0; i < 8; ++i) r[8 + i] = (_Float16)p[16 + i];
  return r;
}

__device__ __forceinline__ v8f wmma_f16(v16h a, v16h b, v8f c) {
  return __builtin_amdgcn_wmma_f32_16x16x32_f16(false, a, false, b, (short)0, c,
                                                false, false);
}

// ---- gfx1250 async copy: global -> LDS, tracked by ASYNCcnt -----------------
// Each thread issues 4 x B128 covering a contiguous 8KB (2048-float) chunk.
__device__ __forceinline__ void stage_async_chunk(const float* gsrc,
                                                  float* lds_dst, int tid) {
#pragma unroll
  for (int i = 0; i < 4; ++i) {
    int idx = (tid + i * 128) * 4;  // float index, 16B granules
    unsigned loff = (unsigned)(size_t)(lds_dst + idx);  // low 32b = LDS offset
    const float* g = gsrc + idx;
    asm volatile("global_load_async_to_lds_b128 %0, %1, off"
                 :: "v"(loff), "v"(g)
                 : "memory");
  }
}

__device__ __forceinline__ void wait_async_all() {
  asm volatile("s_wait_asynccnt 0x0" ::: "memory");
}

// ---------------------------------------------------------------------------
// Kernel 1: projection GEMM   C_f16[M,64] = X_f32[M,512] * W_f32[512,64]
// block = 128 threads (4 waves); block tile 64 rows x 64 cols; grid.y = head.
// W chunks (32x64 f32, contiguous 8KB) are double-buffered into LDS via
// global_load_async_to_lds_b128, then converted/transposed to f16 [n][k].
// MODE 0: out row-major [m][64];  MODE 1: transposed per batch [b][64][1024]
// (used for V so attn@V B-fragments load directly from global).
// ---------------------------------------------------------------------------
template <int MODE>
__global__ void __launch_bounds__(128)
proj_gemm_kernel(const float* __restrict__ X, const float* __restrict__ W,
                 _Float16* __restrict__ out, int whstride, int ohstride,
                 float scale) {
  __shared__ __align__(16) float Wstage[2][2048];  // raw f32 [kk][nn] chunks
  __shared__ _Float16 Wsh[64][40];                 // f16 [n][k], padded rows
  const int h = blockIdx.y;
  const float* Wh = W + (size_t)h * whstride;
  _Float16* outh = out + (size_t)h * ohstride;
  const int tid = threadIdx.x;
  const int wave = tid >> 5;
  const int lane = tid & 31;
  const int m0 = blockIdx.x * 64 + wave * 16;

  v8f acc[4];
#pragma unroll
  for (int t = 0; t < 4; ++t)
#pragma unroll
    for (int j = 0; j < 8; ++j) acc[t][j] = 0.0f;

  // Prime the async pipeline with chunk 0.
  stage_async_chunk(Wh, &Wstage[0][0], tid);

  int buf = 0;
  for (int kc = 0; kc < DM; kc += 32) {
    wait_async_all();   // this wave's copies landed in LDS
    __syncthreads();    // join all waves' copies
    if (kc + 32 < DM)   // stream the next chunk into the other buffer
      stage_async_chunk(Wh + (size_t)(kc + 32) * DKN, &Wstage[buf ^ 1][0], tid);

    // convert/transpose staged chunk: Wstage[kk*64+nn] -> Wsh[nn][kk]
#pragma unroll
    for (int i = 0; i < 16; ++i) {
      int e = tid + i * 128;  // 0..2047
      int kk = e >> 6, nn = e & 63;
      Wsh[nn][kk] = (_Float16)Wstage[buf][e];
    }
    __syncthreads();

    v16h a = load_frag_f32(X, m0, DM, kc);
#pragma unroll
    for (int t = 0; t < 4; ++t) {
      v16h b = load_frag_h(&Wsh[0][0], t * 16, 40, 0);
      acc[t] = wmma_f16(a, b, acc[t]);
    }
    __syncthreads();  // Wsh reused next iteration
    buf ^= 1;
  }

  const int mr = (lane >> 4) << 3;
  const int n = lane & 15;
#pragma unroll
  for (int t = 0; t < 4; ++t) {
    int nn = t * 16 + n;
#pragma unroll
    for (int j = 0; j < 8; ++j) {
      int m = m0 + mr + j;
      _Float16 val = (_Float16)(acc[t][j] * scale);
      if (MODE == 0) {
        outh[(size_t)m * DKN + nn] = val;
      } else {  // [b][dk][s]
        int b = m >> 10, s = m & 1023;
        outh[((size_t)b << 16) + ((size_t)nn << 10) + s] = val;
      }
    }
  }
}

// ---------------------------------------------------------------------------
// Kernel 2: logits[h,b,q,k] = (qs . ks) (+ masks)  -> attn buffer (f32)
// qs already carries the 1/sqrt(d_k) scale.  Fully-masked causal tiles skip
// the GEMM entirely and just write NEG.
// ---------------------------------------------------------------------------
__global__ void __launch_bounds__(128)
logits_kernel(const _Float16* __restrict__ qs, const _Float16* __restrict__ ks,
              const float* __restrict__ pad, float* __restrict__ attn) {
  const int b = blockIdx.y, h = blockIdx.z;
  const int tx = blockIdx.x & 15;  // col block (64 cols)
  const int ty = blockIdx.x >> 4;  // row block (64 rows)
  const int wave = threadIdx.x >> 5;
  const int lane = threadIdx.x & 31;
  const int m0 = ty * 64 + wave * 16;
  const int n0 = tx * 64;
  const _Float16* Q = qs + ((size_t)h * MROWS + (size_t)b * SN) * DKN;
  const _Float16* K = ks + ((size_t)h * MROWS + (size_t)b * SN) * DKN;
  float* A = attn + ((size_t)h * BN + b) * SN * SN;
  const int mr = (lane >> 4) << 3;
  const int n = lane & 15;

  // Hoist the two A-fragments (k-chunks of 32 over d_k=64).
  v16h aq0 = load_frag_h(Q, m0, DKN, 0);
  v16h aq1 = load_frag_h(Q, m0, DKN, 32);

#pragma unroll
  for (int t = 0; t < 4; ++t) {
    const int nt0 = n0 + t * 16;
    if (nt0 > m0 + 15) {  // fully above the diagonal: all masked
#pragma unroll
      for (int j = 0; j < 8; ++j)
        A[(size_t)(m0 + mr + j) * SN + nt0 + n] = NEGV;
      continue;  // wave-uniform branch: EXEC stays all-ones around wmma
    }
    v8f acc;
#pragma unroll
    for (int j = 0; j < 8; ++j) acc[j] = 0.0f;
    acc = wmma_f16(aq0, load_frag_h(K, nt0, DKN, 0), acc);
    acc = wmma_f16(aq1, load_frag_h(K, nt0, DKN, 32), acc);

    const int ki = nt0 + n;
    const float pm = pad[(size_t)b * SN + ki] * NEGV;
#pragma unroll
    for (int j = 0; j < 8; ++j) {
      int qi = m0 + mr + j;
      float v = acc[j] + pm;
      if (ki > qi) v = NEGV;  // causal mask
      A[(size_t)qi * SN + ki] = v;
    }
  }
}

// ---------------------------------------------------------------------------
// Kernel 3: in-place row softmax over the last dim (1024) of attn.
// 256 threads / row; float4 per thread; wave32 shfl_xor + LDS reduction.
// ---------------------------------------------------------------------------
__global__ void __launch_bounds__(256)
softmax_kernel(float* __restrict__ attn) {
  float* p = attn + (size_t)blockIdx.x * SN;
  const int t = threadIdx.x;
  const int wv = t >> 5, ln = t & 31;
  __shared__ float red[8];

  float4 v = ((const float4*)p)[t];
  float m = fmaxf(fmaxf(v.x, v.y), fmaxf(v.z, v.w));
#pragma unroll
  for (int off = 16; off > 0; off >>= 1) m = fmaxf(m, __shfl_xor(m, off, 32));
  if (ln == 0) red[wv] = m;
  __syncthreads();
  float rowmax = red[0];
#pragma unroll
  for (int i = 1; i < 8; ++i) rowmax = fmaxf(rowmax, red[i]);
  __syncthreads();

  v.x = __expf(v.x - rowmax);
  v.y = __expf(v.y - rowmax);
  v.z = __expf(v.z - rowmax);
  v.w = __expf(v.w - rowmax);
  float s = v.x + v.y + v.z + v.w;
#pragma unroll
  for (int off = 16; off > 0; off >>= 1) s += __shfl_xor(s, off, 32);
  if (ln == 0) red[wv] = s;
  __syncthreads();
  float rowsum = red[0];
#pragma unroll
  for (int i = 1; i < 8; ++i) rowsum += red[i];

  float inv = 1.0f / rowsum;
  v.x *= inv; v.y *= inv; v.z *= inv; v.w *= inv;
  ((float4*)p)[t] = v;
}

// ---------------------------------------------------------------------------
// Kernel 4: headsum[b,s,dk] = (1/H) * sum_h  attn[h,b,s,:] @ vs[b,:,dk]
// A fragments converted f32->f16 from attn; B fragments loaded directly from
// the transposed V projection vsT[b][dk][s].  k-loop clipped at the causal
// diagonal (attn is exactly 0 above it).  Output f16 with the 1/8 mean folded.
// ---------------------------------------------------------------------------
__global__ void __launch_bounds__(128)
attnv_kernel(const float* __restrict__ attn, const _Float16* __restrict__ vsT,
             _Float16* __restrict__ hs) {
  const int b = blockIdx.y;
  const int wave = threadIdx.x >> 5;
  const int lane = threadIdx.x & 31;
  const int m0 = blockIdx.x * 64 + wave * 16;  // query row within batch
  const _Float16* V = vsT + (size_t)b * (DKN * SN);

  v8f acc[4];
#pragma unroll
  for (int t = 0; t < 4; ++t)
#pragma unroll
    for (int j = 0; j < 8; ++j) acc[t][j] = 0.0f;

  // attn[m][k] == 0 for k > m, so only k0 <= m0 contributes for this wave.
  for (int k0 = 0; k0 < m0 + 16; k0 += 32) {
    v16h bf[4];
#pragma unroll
    for (int t = 0; t < 4; ++t) bf[t] = load_frag_h(V, t * 16, SN, k0);
#pragma unroll
    for (int h = 0; h < HN; ++h) {
      const float* Ah = attn + ((size_t)h * BN + b) * SN * SN;
      __builtin_prefetch(Ah + (size_t)m0 * SN + k0 + 32, 0, 0);
      v16h a = load_frag_f32(Ah, m0, SN, k0);
#pragma unroll
      for (int t = 0; t < 4; ++t) acc[t] = wmma_f16(a, bf[t], acc[t]);
    }
  }

  const int mr = (lane >> 4) << 3;
  const int n = lane & 15;
#pragma unroll
  for (int t = 0; t < 4; ++t)
#pragma unroll
    for (int j = 0; j < 8; ++j)
      hs[(size_t)(b * SN + m0 + mr + j) * DKN + t * 16 + n] =
          (_Float16)(acc[t][j] * 0.125f);  // mean over H=8 heads
}

// ---------------------------------------------------------------------------
// Kernel 5: out[M,512] = headsum_f16[M,64] @ Wo_f32[64,512]   (f32 out)
// ---------------------------------------------------------------------------
__global__ void __launch_bounds__(128)
outproj_kernel(const _Float16* __restrict__ hs, const float* __restrict__ Wo,
               float* __restrict__ out) {
  __shared__ _Float16 Wsh[64][40];
  const int n0b = blockIdx.y * 64;
  const int wave = threadIdx.x >> 5;
  const int lane = threadIdx.x & 31;
  const int m0 = blockIdx.x * 64 + wave * 16;

  v8f acc[4];
#pragma unroll
  for (int t = 0; t < 4; ++t)
#pragma unroll
    for (int j = 0; j < 8; ++j) acc[t][j] = 0.0f;

#pragma unroll
  for (int kc = 0; kc < DKN; kc += 32) {
#pragma unroll
    for (int i = 0; i < 16; ++i) {
      int e = threadIdx.x + i * 128;
      int kk = e >> 6, nn = e & 63;
      Wsh[nn][kk] = (_Float16)Wo[(size_t)(kc + kk) * DM + n0b + nn];
    }
    __syncthreads();
    v16h a = load_frag_h(hs, m0, DKN, kc);
#pragma unroll
    for (int t = 0; t < 4; ++t) {
      v16h bfr = load_frag_h(&Wsh[0][0], t * 16, 40, 0);
      acc[t] = wmma_f16(a, bfr, acc[t]);
    }
    __syncthreads();
  }

  const int mr = (lane >> 4) << 3;
  const int n = lane & 15;
#pragma unroll
  for (int t = 0; t < 4; ++t)
#pragma unroll
    for (int j = 0; j < 8; ++j)
      out[(size_t)(m0 + mr + j) * DM + n0b + t * 16 + n] = acc[t][j];
}

// ---------------------------------------------------------------------------
extern "C" void kernel_launch(void* const* d_in, const int* in_sizes, int n_in,
                              void* d_out, int out_size, void* d_ws,
                              size_t ws_size, hipStream_t stream) {
  const float* q   = (const float*)d_in[0];
  const float* k   = (const float*)d_in[1];
  const float* v   = (const float*)d_in[2];
  // d_in[3] = causal_mask (handled analytically: ki > qi)
  const float* pad = (const float*)d_in[4];
  const float* Wq  = (const float*)d_in[5];
  const float* Wk  = (const float*)d_in[6];
  const float* Wv  = (const float*)d_in[7];
  const float* Wo  = (const float*)d_in[8];

  float* out  = (float*)d_out;
  float* attn = out + (size_t)BN * SN * DM;  // outputs concatenated flat

  char* ws = (char*)d_ws;
  _Float16* qs  = (_Float16*)(ws);                        // 16 MB  [h][m][64]
  _Float16* ks  = (_Float16*)(ws + ((size_t)16 << 20));   // 16 MB  [h][m][64]
  _Float16* vsT = (_Float16*)(ws + ((size_t)32 << 20));   //  2 MB  [b][64][s]
  _Float16* hs  = (_Float16*)(ws + ((size_t)34 << 20));   //  2 MB  [m][64]

  dim3 blk(128);
  // Q/K per-head projections (scale 1/sqrt(64) folded into Q), V shared proj.
  proj_gemm_kernel<0><<<dim3(MROWS / 64, HN), blk, 0, stream>>>(
      q, Wq, qs, DM * DKN, MROWS * DKN, 0.125f);
  proj_gemm_kernel<0><<<dim3(MROWS / 64, HN), blk, 0, stream>>>(
      k, Wk, ks, DM * DKN, MROWS * DKN, 1.0f);
  proj_gemm_kernel<1><<<dim3(MROWS / 64, 1), blk, 0, stream>>>(
      v, Wv, vsT, 0, 0, 1.0f);

  logits_kernel<<<dim3((SN / 64) * (SN / 64), BN, HN), blk, 0, stream>>>(
      qs, ks, pad, attn);

  softmax_kernel<<<dim3(HN * BN * SN), dim3(256), 0, stream>>>(attn);

  attnv_kernel<<<dim3(SN / 64, BN), blk, 0, stream>>>(attn, vsT, hs);

  outproj_kernel<<<dim3(MROWS / 64, DM / 64), blk, 0, stream>>>(hs, Wo, out);
}
